// HGWaveNet_30124900614689
// MI455X (gfx1250) — matrix operand order repaired and checked by reading, not compiled, not run on hardware
//
#include <hip/hip_runtime.h>
#include <hip/hip_bf16.h>
#include <stdint.h>

typedef __attribute__((ext_vector_type(16))) _Float16 v16h;
typedef __attribute__((ext_vector_type(8)))  float    v8f;

#define DFEAT 128
#define EPSN     1e-15f
#define BALL_EPS 1e-5f

__device__ __forceinline__ float wave_allsum(float v) {
#pragma unroll
  for (int off = 16; off > 0; off >>= 1) v += __shfl_xor(v, off, 32);
  return v;
}

// ---------------------------------------------------------------------------
// Kernel 1: logmap0 on node_embeddings + the 3 history matrices, emit f16.
// One wave per row (wave32, 4 floats/lane), wave-reduce the squared norm.
// Af16 : [N,128]  tangent(node_embeddings)
// THf16: [N,384]  j = k*128 + i ordering (hist-k block-contiguous)
// ---------------------------------------------------------------------------
__global__ __launch_bounds__(256) void prep_logmap_f16(
    const float* __restrict__ emb, const float* __restrict__ h1,
    const float* __restrict__ h2,  const float* __restrict__ h3,
    const float* __restrict__ curv,
    _Float16* __restrict__ Af16, _Float16* __restrict__ THf16, int n_rows) {
  const int gw   = blockIdx.x * (blockDim.x >> 5) + (threadIdx.x >> 5);
  const int lane = threadIdx.x & 31;
  if (gw >= 4 * n_rows) return;
  const int which = gw / n_rows;
  const int n     = gw - which * n_rows;
  const float* src = (which == 0) ? emb : (which == 1) ? h1 : (which == 2) ? h2 : h3;
  const float4 x = ((const float4*)src)[(size_t)n * (DFEAT / 4) + lane];
  float ss = wave_allsum(x.x * x.x + x.y * x.y + x.z * x.z + x.w * x.w);
  const float c  = fabsf(curv[0]);
  const float sc = sqrtf(c);
  float xn  = fmaxf(sqrtf(ss), EPSN);
  float arg = fminf(fmaxf(sc * xn, -1.f + BALL_EPS), 1.f - BALL_EPS);
  float at  = 0.5f * __logf((1.f + arg) / (1.f - arg));  // artanh
  float s   = at / (sc * xn);
  _Float16* dst = (which == 0) ? (Af16 + (size_t)n * DFEAT)
                               : (THf16 + (size_t)n * (3 * DFEAT) + (which - 1) * DFEAT);
  union { _Float16 h[4]; uint2 u; } p;
  p.h[0] = (_Float16)(s * x.x); p.h[1] = (_Float16)(s * x.y);
  p.h[2] = (_Float16)(s * x.z); p.h[3] = (_Float16)(s * x.w);
  ((uint2*)dst)[lane] = p.u;
}

// ---------------------------------------------------------------------------
// Kernel 2: pack weights into a WMMA B-fragment layout (contiguous k order,
// 16-bit B 32x16: lanes 0-15 hold K 0..15 of the 32-chunk, lanes 16-31 hold
// K 16..31; VGPR pair j = (K=2j, K=2j+1)).  One thread per packed f16 pair.
// mode 0: B(k,n) = lin_w[n*128 + k]   (K=128)
// mode 1: B(j,n) = conv_w[n*384 + (j%128)*3 + (j/128)]  (K=384, j=k*128+i)
// ---------------------------------------------------------------------------
__global__ void pack_B(const float* __restrict__ W, unsigned* __restrict__ Bp,
                       int K, int mode) {
  const int t = blockIdx.x * blockDim.x + threadIdx.x;
  if (t >= K * 64) return;
  const int j    = t & 7;
  const int lane = (t >> 3) & 31;
  const int frag = t >> 8;               // kt*8 + ct
  const int ct = frag & 7, kt = frag >> 3;
  const int nn = ct * 16 + (lane & 15);
  const int k0 = kt * 32 + ((lane >= 16) ? 16 : 0) + 2 * j;
  float v0, v1;
  if (mode == 0) {
    v0 = W[nn * 128 + k0];
    v1 = W[nn * 128 + k0 + 1];
  } else {
    const int i0 = k0 & 127, kk0 = k0 >> 7;
    const int i1 = (k0 + 1) & 127, kk1 = (k0 + 1) >> 7;
    v0 = W[nn * 384 + i0 * 3 + kk0];
    v1 = W[nn * 384 + i1 * 3 + kk1];
  }
  union { _Float16 h[2]; unsigned u; } p;
  p.h[0] = (_Float16)v0; p.h[1] = (_Float16)v1;
  Bp[t] = p.u;
}

// ---------------------------------------------------------------------------
// Kernel 3: WMMA GEMM  C[M,128] = A[M,K](f16) * B(K,128) + bias, f32 accum.
// Block = 8 waves; wave w owns row-tile (blockIdx*8+w): 16 rows x 128 cols
// = 8 accumulators of v8f.  B fragments staged via LDS (32 KB slab / stage).
// A fragment in contiguous k order, as pack_B: lane<16 holds K 0..15,
// lane>=16 holds K 16..31 -> two b128 loads per lane.
// ---------------------------------------------------------------------------
template <int K>
__global__ __launch_bounds__(256) void wmma_gemm_f16(
    const _Float16* __restrict__ A, const unsigned* __restrict__ Bp,
    const float* __restrict__ bias, float* __restrict__ C, int M) {
  __shared__ uint4 bsh[2048];                       // 32 KB = 4 kt-groups
  constexpr int numKT  = K / 32;
  constexpr int stages = numKT / 4;                 // K=128 -> 1, K=384 -> 3
  const int wave = threadIdx.x >> 5;
  const int lane = threadIdx.x & 31;
  const int rt   = blockIdx.x * 8 + wave;
  const int tilesM = M >> 4;
  const int l16  = (lane >= 16) ? 1 : 0;
  const int row  = (rt << 4) + (lane & 15);
  v8f acc[8] = {};
#pragma unroll
  for (int st = 0; st < stages; ++st) {
    const uint4* src = (const uint4*)Bp + st * 2048;
    for (int q = threadIdx.x; q < 2048; q += 256) bsh[q] = src[q];
    __syncthreads();
    if (rt < tilesM) {                              // wave-uniform: EXEC all-1s
#pragma unroll
      for (int kl = 0; kl < 4; ++kl) {
        const int kt = st * 4 + kl;
        const _Float16* ap = A + (size_t)row * K + kt * 32 + (l16 ? 16 : 0);
        union { uint4 q[2]; v16h v; } af;
        af.q[0] = *(const uint4*)ap;                // K +0..7   (global_load_b128)
        af.q[1] = *(const uint4*)(ap + 8);          // K +8..15
#pragma unroll
        for (int ct = 0; ct < 8; ++ct) {
          union { uint4 q[2]; v16h v; } bf;
          const int base = ((kl * 8 + ct) * 32 + lane) * 2;
          bf.q[0] = bsh[base];                      // ds_load_b128
          bf.q[1] = bsh[base + 1];
          acc[ct] = __builtin_amdgcn_wmma_f32_16x16x32_f16(
              false, af.v, false, bf.v, (short)0, acc[ct], false, false);
        }
      }
    }
    __syncthreads();
  }
  if (rt < tilesM) {
    const int rbase = (rt << 4) + (l16 ? 8 : 0);    // C/D layout: lanes 16-31 = M+8
#pragma unroll
    for (int ct = 0; ct < 8; ++ct) {
      const int col = ct * 16 + (lane & 15);
      const float b = bias[col];
#pragma unroll
      for (int r = 0; r < 8; ++r)
        C[(size_t)(rbase + r) * DFEAT + col] = acc[ct][r] + b;
    }
  }
}

// ---------------------------------------------------------------------------
// Kernel 4: zero-fill scratch (graph-capture safe).
// ---------------------------------------------------------------------------
__global__ void zero_f32(float* __restrict__ p, int n) {
  const int t = blockIdx.x * blockDim.x + threadIdx.x;
  if (t < n) p[t] = 0.f;
}

// ---------------------------------------------------------------------------
// Kernel 5: edge gather + segment-sum.  One wave per edge; lane carries 4
// floats.  transformed (25.6 MB) is L2-resident; adds are no-return
// global_atomic_add_f32 at agent scope.
// ---------------------------------------------------------------------------
__global__ __launch_bounds__(256) void edge_scatter(
    const float* __restrict__ T, const int* __restrict__ esrc,
    const int* __restrict__ edst, float* __restrict__ summed,
    float* __restrict__ deg, int E) {
  const int w    = blockIdx.x * (blockDim.x >> 5) + (threadIdx.x >> 5);
  const int lane = threadIdx.x & 31;
  if (w >= E) return;
  const int s = esrc[w], d = edst[w];
  const float4 v = ((const float4*)T)[(size_t)s * 32 + lane];
  float* p = summed + (size_t)d * DFEAT + lane * 4;
  __hip_atomic_fetch_add(p + 0, v.x, __ATOMIC_RELAXED, __HIP_MEMORY_SCOPE_AGENT);
  __hip_atomic_fetch_add(p + 1, v.y, __ATOMIC_RELAXED, __HIP_MEMORY_SCOPE_AGENT);
  __hip_atomic_fetch_add(p + 2, v.z, __ATOMIC_RELAXED, __HIP_MEMORY_SCOPE_AGENT);
  __hip_atomic_fetch_add(p + 3, v.w, __ATOMIC_RELAXED, __HIP_MEMORY_SCOPE_AGENT);
  if (lane == 0)
    __hip_atomic_fetch_add(deg + d, 1.0f, __ATOMIC_RELAXED, __HIP_MEMORY_SCOPE_AGENT);
}

// ---------------------------------------------------------------------------
// Kernel 6: neigh = summed/deg, then tangent_cur = logmap0(expmap0(neigh)).
// In-place over `buf` (row fully held by one wave).  One norm reduction:
// ||expmap0(u)|| = tanh(sc*||u||)/sc exactly, so both scales are lane-uniform.
// ---------------------------------------------------------------------------
__global__ __launch_bounds__(256) void neigh_round_trip(
    float* __restrict__ buf, const float* __restrict__ deg,
    const float* __restrict__ curv, int n_rows) {
  const int n    = blockIdx.x * (blockDim.x >> 5) + (threadIdx.x >> 5);
  const int lane = threadIdx.x & 31;
  if (n >= n_rows) return;
  const float dg  = deg[n];
  const float inv = (dg > 0.f) ? (1.f / fmaxf(dg, 1.f)) : 0.f;
  float4 u = ((const float4*)buf)[(size_t)n * 32 + lane];
  u.x *= inv; u.y *= inv; u.z *= inv; u.w *= inv;
  float ss = wave_allsum(u.x * u.x + u.y * u.y + u.z * u.z + u.w * u.w);
  const float c  = fabsf(curv[0]);
  const float sc = sqrtf(c);
  float un  = fmaxf(sqrtf(ss), EPSN);
  float es  = tanhf(sc * un) / (sc * un);          // expmap0 scale
  float xn  = fmaxf(es * un, EPSN);                // ||expmap0(u)||
  float arg = fminf(fmaxf(sc * xn, -1.f + BALL_EPS), 1.f - BALL_EPS);
  float at  = 0.5f * __logf((1.f + arg) / (1.f - arg));
  float t   = es * (at / (sc * xn));               // composed scale
  u.x *= t; u.y *= t; u.z *= t; u.w *= t;
  ((float4*)buf)[(size_t)n * 32 + lane] = u;
}

// ---------------------------------------------------------------------------
// Kernel 7: out = expmap0(conv_out + tangent_cur).
// ---------------------------------------------------------------------------
__global__ __launch_bounds__(256) void final_expmap(
    const float* __restrict__ conv, const float* __restrict__ tcur,
    const float* __restrict__ curv, float* __restrict__ out, int n_rows) {
  const int n    = blockIdx.x * (blockDim.x >> 5) + (threadIdx.x >> 5);
  const int lane = threadIdx.x & 31;
  if (n >= n_rows) return;
  float4 a = ((const float4*)conv)[(size_t)n * 32 + lane];
  float4 b = ((const float4*)tcur)[(size_t)n * 32 + lane];
  float4 u = make_float4(a.x + b.x, a.y + b.y, a.z + b.z, a.w + b.w);
  float ss = wave_allsum(u.x * u.x + u.y * u.y + u.z * u.z + u.w * u.w);
  const float c  = fabsf(curv[0]);
  const float sc = sqrtf(c);
  float un = fmaxf(sqrtf(ss), EPSN);
  float s  = tanhf(sc * un) / (sc * un);
  u.x *= s; u.y *= s; u.z *= s; u.w *= s;
  ((float4*)out)[(size_t)n * 32 + lane] = u;
}

// ---------------------------------------------------------------------------
extern "C" void kernel_launch(void* const* d_in, const int* in_sizes, int n_in,
                              void* d_out, int out_size, void* d_ws, size_t ws_size,
                              hipStream_t stream) {
  const float* emb    = (const float*)d_in[0];
  const float* h1     = (const float*)d_in[1];
  const float* h2     = (const float*)d_in[2];
  const float* h3     = (const float*)d_in[3];
  const float* lin_w  = (const float*)d_in[4];
  const float* lin_b  = (const float*)d_in[5];
  const float* conv_w = (const float*)d_in[6];
  const float* conv_b = (const float*)d_in[7];
  const float* curv   = (const float*)d_in[8];
  const int*   esrc   = (const int*)d_in[9];
  const int*   edst   = (const int*)d_in[10];
  const int N = in_sizes[0] / DFEAT;   // 50000 (divisible by 16)
  const int E = in_sizes[9];           // 625000
  float* out = (float*)d_out;

  // Workspace carve-up (all offsets 128B-aligned).
  char* ws = (char*)d_ws;
  _Float16* Af16   = (_Float16*)ws;                              // N*128*2
  _Float16* THf16  = (_Float16*)(ws + (size_t)N * DFEAT * 2);    // N*384*2
  size_t off = (size_t)N * DFEAT * 2 + (size_t)N * 3 * DFEAT * 2;
  unsigned* W1p = (unsigned*)(ws + off);  off += 128 * 64 * 4;   // 32 KB packed
  unsigned* W2p = (unsigned*)(ws + off);  off += 384 * 64 * 4;   // 96 KB packed
  float* buf1 = (float*)(ws + off);       off += (size_t)N * DFEAT * 4;  // transformed / conv_out
  float* buf2 = (float*)(ws + off);       off += (size_t)N * DFEAT * 4;  // summed -> tangent_cur
  float* deg  = (float*)(ws + off);                               // N floats (contiguous after buf2)

  // 1) logmap0 -> f16 operands (4N waves, 8 waves/block)
  prep_logmap_f16<<<(4 * N + 7) / 8, 256, 0, stream>>>(emb, h1, h2, h3, curv,
                                                       Af16, THf16, N);
  // 2) pack weights into WMMA B-fragment layout
  pack_B<<<(128 * 64 + 255) / 256, 256, 0, stream>>>(lin_w, W1p, 128, 0);
  pack_B<<<(384 * 64 + 255) / 256, 256, 0, stream>>>(conv_w, W2p, 384, 1);
  // 3) GEMM1: transformed = tangent @ lin_w.T + lin_b   (K=128)
  wmma_gemm_f16<128><<<(N / 16 + 7) / 8, 256, 0, stream>>>(Af16, W1p, lin_b, buf1, N);
  // 4) zero summed + deg (contiguous)
  zero_f32<<<(N * DFEAT + N + 255) / 256, 256, 0, stream>>>(buf2, N * DFEAT + N);
  // 5) edge gather + atomic segment-sum
  edge_scatter<<<(E + 7) / 8, 256, 0, stream>>>(buf1, esrc, edst, buf2, deg, E);
  // 6) neigh normalize + expmap0/logmap0 round trip (in place)
  neigh_round_trip<<<(N + 7) / 8, 256, 0, stream>>>(buf2, deg, curv, N);
  // 7) GEMM2: conv_out = TH @ W2 + conv_b   (K=384), reuses buf1
  wmma_gemm_f16<384><<<(N / 16 + 7) / 8, 256, 0, stream>>>(THf16, W2p, conv_b, buf1, N);
  // 8) out = expmap0(conv_out + tangent_cur)
  final_expmap<<<(N + 7) / 8, 256, 0, stream>>>(buf1, buf2, curv, out, N);
  (void)n_in; (void)out_size; (void)ws_size;
}
